// LstmConv_21655225106656
// MI455X (gfx1250) — matrix operand order repaired and checked by reading, not compiled, not run on hardware
//
#include <hip/hip_runtime.h>
#include <cmath>

#define N_NODES 50000
#define N_EDGES 800000
#define HIDDEN  128
#define MSG     64
#define KCAT    192      // HIDDEN + MSG (fused GEMM K)
#define NGATES  256      // 4 * MSG

typedef _Float16 v16h __attribute__((ext_vector_type(16)));
typedef _Float16 v8h  __attribute__((ext_vector_type(8)));
typedef float    v8f  __attribute__((ext_vector_type(8)));

// ---------------- workspace layout (bytes) ----------------
constexpr size_t SUM_BYTES  = (size_t)N_NODES * HIDDEN * 4;   // 25,600,000
constexpr size_t SUM0_OFF   = 0;
constexpr size_t SUM1_OFF   = SUM0_OFF + SUM_BYTES;           // 25,600,000
constexpr size_t CNT_BYTES  = 262144;                          // padded 50000*4
constexpr size_t CNT0_OFF   = SUM1_OFF + SUM_BYTES;           // 51,200,000
constexpr size_t CNT1_OFF   = CNT0_OFF + CNT_BYTES;
constexpr size_t GATES_OFF  = 0;                               // aliases sum0+sum1 (51.2 MB)
constexpr size_t AH_OFF     = CNT1_OFF + CNT_BYTES;           // 51,724,288 (32B aligned)
constexpr size_t AH_BYTES   = (size_t)N_NODES * KCAT * 2;     // 19,200,000
constexpr size_t RT_OFF     = AH_OFF + AH_BYTES;              // 70,924,288
constexpr size_t RT_BYTES   = (size_t)N_NODES * MSG * 4;      // 12,800,000
constexpr size_t WCAT_OFF   = RT_OFF + RT_BYTES;              // 83,724,288
constexpr size_t WCAT_BYTES = (size_t)NGATES * KCAT * 2;      // 98,304
constexpr size_t BSUM_OFF   = WCAT_OFF + WCAT_BYTES;
constexpr size_t ZERO_F4    = (AH_OFF) / 16;                   // float4 count to zero

// ---------------- kernels ----------------

__global__ void zero_ws_kernel(float4* ws) {
    size_t i = (size_t)blockIdx.x * blockDim.x + threadIdx.x;
    if (i < ZERO_F4) ws[i] = make_float4(0.f, 0.f, 0.f, 0.f);
}

// one wave per edge: lane handles 4 consecutive features; lane0 bumps the count
__global__ void scatter_mean_kernel(const float* __restrict__ feat,
                                    const int* __restrict__ src,
                                    const int* __restrict__ dst,
                                    float* __restrict__ sum,
                                    float* __restrict__ cnt) {
    int warp = (int)((blockIdx.x * blockDim.x + threadIdx.x) >> 5);
    int lane = threadIdx.x & 31;
    if (warp >= N_EDGES) return;
    int s = src[warp];
    int d = dst[warp];
    float4 v = reinterpret_cast<const float4*>(feat + (size_t)s * HIDDEN)[lane];
    float* p = sum + (size_t)d * HIDDEN + lane * 4;
    atomicAdd(p + 0, v.x);
    atomicAdd(p + 1, v.y);
    atomicAdd(p + 2, v.z);
    atomicAdd(p + 3, v.w);
    if (lane == 0) atomicAdd(cnt + d, 1.0f);
}

// per node: compute rst, pack A = [feat | G_t] as f16, save R_t as f32
__global__ void combine_kernel(const float* __restrict__ feat,
                               const float* __restrict__ sum0,
                               const float* __restrict__ sum1,
                               const float* __restrict__ cnt0,
                               const float* __restrict__ cnt1,
                               _Float16* __restrict__ Ah,
                               float* __restrict__ Rt) {
    int n = blockIdx.x;       // 50000 blocks
    int j = threadIdx.x;      // 128 threads
    float c0 = cnt0[n], c1 = cnt1[n];
    float m0 = sum0[(size_t)n * HIDDEN + j] / fmaxf(c0, 1.0f);
    float m1 = sum1[(size_t)n * HIDDEN + j] / fmaxf(c1, 1.0f);
    float act = fmaxf((c0 > 0.f ? 1.f : 0.f) + (c1 > 0.f ? 1.f : 0.f), 1.0f);
    float r = (m0 + m1) / act;
    Ah[(size_t)n * KCAT + j] = (_Float16)feat[(size_t)n * HIDDEN + j];
    if (j < MSG) Ah[(size_t)n * KCAT + HIDDEN + j] = (_Float16)r;   // G_t
    else         Rt[(size_t)n * MSG + (j - MSG)] = r;               // R_t (f32)
}

// W row n = [W_ih[n,:] | W_hh[n,:]] as f16; bias sum
__global__ void prep_weights_kernel(const float* __restrict__ Wih,
                                    const float* __restrict__ Whh,
                                    const float* __restrict__ bih,
                                    const float* __restrict__ bhh,
                                    _Float16* __restrict__ Wcat,
                                    float* __restrict__ bsum) {
    int n = blockIdx.x;       // 256 blocks
    int j = threadIdx.x;      // 192 threads
    float w = (j < HIDDEN) ? Wih[(size_t)n * HIDDEN + j]
                           : Whh[(size_t)n * MSG + (j - HIDDEN)];
    Wcat[(size_t)n * KCAT + j] = (_Float16)w;
    if (j == 0) bsum[n] = bih[n] + bhh[n];
}

// one wave per 16x16 tile of gates; K=192 -> 6 x v_wmma_f32_16x16x32_f16
__global__ void gemm_wmma_kernel(const _Float16* __restrict__ Ah,
                                 const _Float16* __restrict__ Wcat,
                                 const float* __restrict__ bsum,
                                 float* __restrict__ gates) {
    int wave = threadIdx.x >> 5;
    int lane = threadIdx.x & 31;
    int tile = blockIdx.x * 8 + wave;        // 6250 blocks * 8 waves = 50000 tiles
    int mtile = tile >> 4;                    // 3125 M tiles
    int ntile = tile & 15;                    // 16 N tiles
    int m0 = mtile << 4;
    int n0 = ntile << 4;
    int lr = lane & 15;
    int hi = lane >> 4;                       // 0: lanes 0-15, 1: lanes 16-31

    const _Float16* arow = Ah + (size_t)(m0 + lr) * KCAT;
    const _Float16* brow = Wcat + (size_t)(n0 + lr) * KCAT;
    int a_off0 = hi ? 16 : 0;                 // A: K {0..7 | 16..23}
    int a_off1 = a_off0 + 8;                  //    K {8..15 | 24..31}
    int b_off  = hi ? 16 : 0;                 // B: K {0..15 | 16..31}

    v8f c = {};
#pragma unroll
    for (int k0 = 0; k0 < KCAT; k0 += 32) {
        v8h alo = *reinterpret_cast<const v8h*>(arow + k0 + a_off0);
        v8h ahi = *reinterpret_cast<const v8h*>(arow + k0 + a_off1);
        v16h a = __builtin_shufflevector(alo, ahi,
                                         0, 1, 2, 3, 4, 5, 6, 7,
                                         8, 9, 10, 11, 12, 13, 14, 15);
        v16h b = *reinterpret_cast<const v16h*>(brow + k0 + b_off);
        c = __builtin_amdgcn_wmma_f32_16x16x32_f16(
                /*neg_a=*/false, a, /*neg_b=*/false, b,
                /*c_mod=*/(short)0, c, /*reuse_a=*/false, /*reuse_b=*/false);
    }

    // C/D layout: VGPR r, lanes 0-15 -> (M=r, N=lane); lanes 16-31 -> (M=r+8, N=lane-16)
    float bias = bsum[n0 + lr];
    float* g = gates + (size_t)(m0 + (hi ? 8 : 0)) * NGATES + (n0 + lr);
#pragma unroll
    for (int r = 0; r < 8; ++r) g[(size_t)r * NGATES] = c[r] + bias;
}

__device__ __forceinline__ float sigf(float x) { return 1.0f / (1.0f + expf(-x)); }

__global__ void lstm_elem_kernel(const float* __restrict__ gates,
                                 const float* __restrict__ Rt,
                                 float* __restrict__ out) {
    size_t idx = (size_t)blockIdx.x * blockDim.x + threadIdx.x;  // 50000*64 threads
    int n = (int)(idx >> 6);
    int j = (int)(idx & 63);
    const float* gr = gates + (size_t)n * NGATES;
    float gi = gr[j];
    float gf = gr[MSG + j];
    float gg = gr[2 * MSG + j];
    float go = gr[3 * MSG + j];
    float R  = Rt[(size_t)n * MSG + j];
    float c1 = sigf(gf) * R + sigf(gi) * tanhf(gg);
    float h1 = sigf(go) * tanhf(c1);
    out[(size_t)n * HIDDEN + j]       = h1;
    out[(size_t)n * HIDDEN + MSG + j] = c1;
}

// ---------------- launcher ----------------

extern "C" void kernel_launch(void* const* d_in, const int* in_sizes, int n_in,
                              void* d_out, int out_size, void* d_ws, size_t ws_size,
                              hipStream_t stream) {
    const float* feat = (const float*)d_in[0];
    const int*   src0 = (const int*)d_in[1];
    const int*   dst0 = (const int*)d_in[2];
    const int*   src1 = (const int*)d_in[3];
    const int*   dst1 = (const int*)d_in[4];
    const float* Wih  = (const float*)d_in[5];
    const float* Whh  = (const float*)d_in[6];
    const float* bih  = (const float*)d_in[7];
    const float* bhh  = (const float*)d_in[8];
    float* out = (float*)d_out;

    char* ws = (char*)d_ws;
    float*    sum0  = (float*)(ws + SUM0_OFF);
    float*    sum1  = (float*)(ws + SUM1_OFF);
    float*    cnt0  = (float*)(ws + CNT0_OFF);
    float*    cnt1  = (float*)(ws + CNT1_OFF);
    float*    gates = (float*)(ws + GATES_OFF);   // aliases sum0/sum1 (consumed by then)
    _Float16* Ah    = (_Float16*)(ws + AH_OFF);
    float*    Rt    = (float*)(ws + RT_OFF);
    _Float16* Wcat  = (_Float16*)(ws + WCAT_OFF);
    float*    bsum  = (float*)(ws + BSUM_OFF);

    // 1) zero accumulation buffers (12628 blocks * 256 * float4 == ZERO_F4)
    zero_ws_kernel<<<(unsigned)((ZERO_F4 + 255) / 256), 256, 0, stream>>>((float4*)d_ws);

    // 2) scatter-sum both edge types (one wave per edge)
    const unsigned scat_blocks = (N_EDGES * 32 + 255) / 256;   // 100000
    scatter_mean_kernel<<<scat_blocks, 256, 0, stream>>>(feat, src0, dst0, sum0, cnt0);
    scatter_mean_kernel<<<scat_blocks, 256, 0, stream>>>(feat, src1, dst1, sum1, cnt1);

    // 3) means -> rst; pack A=[feat|G_t] f16, R_t f32
    combine_kernel<<<N_NODES, HIDDEN, 0, stream>>>(feat, sum0, sum1, cnt0, cnt1, Ah, Rt);

    // 4) weight concat + bias sum
    prep_weights_kernel<<<NGATES, KCAT, 0, stream>>>(Wih, Whh, bih, bhh, Wcat, bsum);

    // 5) fused GEMM via WMMA: gates = A @ Wcat^T + b   (overwrites sum region)
    gemm_wmma_kernel<<<6250, 256, 0, stream>>>(Ah, Wcat, bsum, gates);

    // 6) LSTM elementwise -> out = [h1 | c1]
    lstm_elem_kernel<<<(N_NODES * MSG) / 256, 256, 0, stream>>>(gates, Rt, out);
}
